// Low_Rank_WindowAttention_81080392614565
// MI455X (gfx1250) — matrix operand order repaired and checked by reading, not compiled, not run on hardware
//
#include <hip/hip_runtime.h>
#include <hip/hip_bf16.h>

typedef __attribute__((ext_vector_type(16))) __bf16 v16bf;
typedef __attribute__((ext_vector_type(8)))  __bf16 v8bf;
typedef __attribute__((ext_vector_type(8)))  float  v8f;

// ---------------------------------------------------------------------------
// gfx1250 async copy: global -> LDS without VGPR round trip (ASYNCcnt).
// Per-lane 16B transfer; callers keep EXEC full (uniform chunk counts).
// ---------------------------------------------------------------------------
static __device__ __forceinline__ void async_copy_b128(void* lds, const void* g) {
  unsigned l = (unsigned)(size_t)lds;   // generic LDS ptr: low 32 bits = LDS addr
  asm volatile("global_load_async_to_lds_b128 %0, %1, off"
               :: "v"(l), "v"(g) : "memory");
}
// Wait until all async transfers of this wave are done.
static __device__ __forceinline__ void wait_async0() {
  asm volatile("s_wait_asynccnt 0" ::: "memory");
}
// Double-buffer wait: allow the 3 most recently issued transfers (next tile)
// to stay in flight; async loads complete in order, so the previous tile's
// 3 transfers are guaranteed complete after this.
static __device__ __forceinline__ void wait_async3() {
  asm volatile("s_wait_asynccnt 3" ::: "memory");
}

// ---------------------------------------------------------------------------
// WMMA fragment helpers (wave32; this file's contiguous k order, same for A and B)
// A (16x32 MxK): lane l row m=l&15; lanes 0-15 K=0..15, lanes 16-31
//   K=16..31.  B (32x16 KxN): lane col n=l&15; lanes 0-15 K=0..15,
//   lanes 16-31 K=16..31.  C/D f32: VGPR r -> m = r + 8*(lane>=16), n=lane&15.
// ---------------------------------------------------------------------------
static __device__ __forceinline__ v16bf load_frag_a(const __bf16* __restrict__ s,
                                                    int ld, int lane) {
  int m  = lane & 15;
  int kb = (lane >> 4) << 4;
  const __bf16* p = s + m * ld + kb;
  v8bf lo = *(const v8bf*)p;
  v8bf hi = *(const v8bf*)(p + 8);
  v16bf a;
#pragma unroll
  for (int e = 0; e < 8; ++e) { a[e] = lo[e]; a[e + 8] = hi[e]; }
  return a;
}

static __device__ __forceinline__ v16bf load_frag_b_nk(const __bf16* __restrict__ s,
                                                       int ld, int lane) {
  int n  = lane & 15;
  int kb = (lane >> 4) << 4;
  const __bf16* p = s + n * ld + kb;
  v8bf lo = *(const v8bf*)p;
  v8bf hi = *(const v8bf*)(p + 8);
  v16bf b;
#pragma unroll
  for (int e = 0; e < 8; ++e) { b[e] = lo[e]; b[e + 8] = hi[e]; }
  return b;
}

static __device__ __forceinline__ v16bf load_frag_b_kn(const __bf16* __restrict__ s,
                                                       int ld, int lane) {
  int n  = lane & 15;
  int kb = (lane >> 4) << 4;
  v16bf b;
#pragma unroll
  for (int e = 0; e < 16; ++e) b[e] = s[(kb + e) * ld + n];
  return b;
}

static __device__ __forceinline__ v8f wmma_bf16(v16bf a, v16bf b, v8f c) {
  return __builtin_amdgcn_wmma_f32_16x16x32_bf16(false, a, false, b, (short)0, c,
                                                 false, false);
}

// ---------------------------------------------------------------------------
// Kernel 0: fp32 -> bf16 convert (x, qkv_w, proj_w) so every GEMM tile fill
// becomes a pure bf16 copy (async-load-to-LDS eligible).
// ---------------------------------------------------------------------------
__global__ void __launch_bounds__(256) cvt_bf16_kernel(const float* __restrict__ src,
                                                       __bf16* __restrict__ dst,
                                                       int n4) {
  int i = blockIdx.x * 256 + threadIdx.x;
  if (i < n4) {
    float4 val = ((const float4*)src)[i];
    __bf16* d = dst + (size_t)i * 4;
    d[0] = (__bf16)val.x; d[1] = (__bf16)val.y;
    d[2] = (__bf16)val.z; d[3] = (__bf16)val.w;
  }
}

// ---------------------------------------------------------------------------
// Kernel 1: qkv = xb @ wqb^T + qkv_b ; scatter bf16 q/k/v [B,H,N,D], q scaled.
// M=100352, K=384, N=1152.  256 thr = 8 waves (4x2), tile 128x64, K-step 32,
// double-buffered async LDS staging (compute tile k overlaps load of k+1).
// ---------------------------------------------------------------------------
__global__ void __launch_bounds__(256) qkv_gemm_kernel(
    const __bf16* __restrict__ xb, const __bf16* __restrict__ wqb,
    const float* __restrict__ qkv_b,
    __bf16* __restrict__ qo, __bf16* __restrict__ ko, __bf16* __restrict__ vo) {
  __shared__ __align__(16) __bf16 sA[2][128 * 32];
  __shared__ __align__(16) __bf16 sB[2][64 * 32];

  const int tid  = threadIdx.x;
  const int lane = tid & 31;
  const int w    = tid >> 5;
  const int wm   = w >> 1;
  const int wn   = w & 1;
  const int m0   = blockIdx.x * 128;
  const int n0   = blockIdx.y * 64;

  // Per-thread fixed chunk coordinates (3 async issues per tile).
  const int ra0 = tid >> 2,          ca0 = (tid & 3) << 3;          // A chunk 0
  const int ra1 = (tid + 256) >> 2,  ca1 = ca0;                     // A chunk 1
  const int rb  = tid >> 2,          cb  = (tid & 3) << 3;          // B chunk

  auto issue = [&](int kt, int buf) {
    const int k0 = kt * 32;
    async_copy_b128(&sA[buf][ra0 * 32 + ca0],
                    xb + (size_t)(m0 + ra0) * 384 + k0 + ca0);
    async_copy_b128(&sA[buf][ra1 * 32 + ca1],
                    xb + (size_t)(m0 + ra1) * 384 + k0 + ca1);
    async_copy_b128(&sB[buf][rb * 32 + cb],
                    wqb + (size_t)(n0 + rb) * 384 + k0 + cb);
  };

  v8f acc[2][2] = {};

  issue(0, 0);
  for (int kt = 0; kt < 12; ++kt) {
    const int cur = kt & 1;
    if (kt + 1 < 12) { issue(kt + 1, cur ^ 1); wait_async3(); }
    else             { wait_async0(); }
    __syncthreads();
    v16bf a0 = load_frag_a(&sA[cur][(wm * 32) * 32], 32, lane);
    v16bf a1 = load_frag_a(&sA[cur][(wm * 32 + 16) * 32], 32, lane);
    v16bf b0 = load_frag_b_nk(&sB[cur][(wn * 32) * 32], 32, lane);
    v16bf b1 = load_frag_b_nk(&sB[cur][(wn * 32 + 16) * 32], 32, lane);
    acc[0][0] = wmma_bf16(a0, b0, acc[0][0]);
    acc[0][1] = wmma_bf16(a0, b1, acc[0][1]);
    acc[1][0] = wmma_bf16(a1, b0, acc[1][0]);
    acc[1][1] = wmma_bf16(a1, b1, acc[1][1]);
    __syncthreads();   // all waves done reading buf[cur] before it is refilled
  }

  // Epilogue: part/head/dest uniform per (wave, j) subtile (384%64==0,
  // heads 32-wide, subtiles 16-aligned) -> no per-lane div/branches on n.
  const int lm = lane & 15;
  const int lh = (lane >> 4) << 3;
#pragma unroll
  for (int j = 0; j < 2; ++j) {
    const int ncol = n0 + wn * 32 + j * 16;
    const int part = ncol / 384;                    // 0:q 1:k 2:v (uniform)
    const int rem  = ncol - part * 384;
    const int h    = rem >> 5;
    const int d    = (rem & 31) + lm;
    __bf16* const dst = (part == 0) ? qo : (part == 1) ? ko : vo;
    const float mul   = (part == 0) ? 0.17677669529663687f : 1.0f;  // 32^-0.5
    const float bias  = qkv_b[ncol + lm];
#pragma unroll
    for (int i = 0; i < 2; ++i) {
#pragma unroll
      for (int r = 0; r < 8; ++r) {
        int m   = m0 + wm * 32 + i * 16 + lh + r;
        int bw  = m / 49;
        int tok = m - bw * 49;
        dst[((bw * 12 + h) * 49 + tok) * 32 + d] =
            (__bf16)((acc[i][j][r] + bias) * mul);
      }
    }
  }
}

// ---------------------------------------------------------------------------
// Kernel 2: per (window, head) attention, 49 tokens padded to 64.
// 128 thr = 4 waves; async-stage Q/K/V, S=QK^T, bias+softmax, O=PV.
// ---------------------------------------------------------------------------
__global__ void __launch_bounds__(128) attn_kernel(
    const __bf16* __restrict__ q, const __bf16* __restrict__ k,
    const __bf16* __restrict__ v, const float* __restrict__ bias_table,
    const int* __restrict__ rel_index, __bf16* __restrict__ aout) {
  __shared__ __align__(16) __bf16 sQ[64 * 32];
  __shared__ __align__(16) __bf16 sK[64 * 32];
  __shared__ __align__(16) __bf16 sV[64 * 32];
  __shared__ __align__(16) float  sS[64 * 64];
  __shared__ __align__(16) __bf16 sP[64 * 64];

  const int tid  = threadIdx.x;
  const int lane = tid & 31;
  const int w    = tid >> 5;
  const int bh   = blockIdx.x;            // b*12 + h
  const int b    = bh / 12;
  const int h    = bh - b * 12;
  const size_t base = (size_t)bh * 49 * 32;

  // Async-stage full padded 64x32 tiles (256 x 16B chunks each, 2/thread).
  // Over-read past row 49 stays inside the workspace (buffers are adjacent).
#pragma unroll
  for (int i = 0; i < 2; ++i) {
    int off = (tid + 128 * i) << 3;       // bf16 element offset, 16B chunks
    async_copy_b128(sQ + off, q + base + off);
    async_copy_b128(sK + off, k + base + off);
    async_copy_b128(sV + off, v + base + off);
  }
  wait_async0();
  __syncthreads();                         // all async LDS writes visible
  // Zero pad rows 49..63 (V must be clean; Q/K for robustness).
  {
    unsigned* uQ = (unsigned*)sQ;
    unsigned* uK = (unsigned*)sK;
    unsigned* uV = (unsigned*)sV;
    for (int i = 784 + tid; i < 1024; i += 128) { uQ[i] = 0u; uK[i] = 0u; uV[i] = 0u; }
  }
  __syncthreads();

  // S = Q K^T : wave w owns rows 16w..16w+15; 4 column tiles, K-dim 32.
  {
    v16bf aq = load_frag_a(sQ + w * 16 * 32, 32, lane);
#pragma unroll
    for (int j = 0; j < 4; ++j) {
      v16bf bk = load_frag_b_nk(sK + j * 16 * 32, 32, lane);
      v8f z = {};
      v8f s = wmma_bf16(aq, bk, z);
#pragma unroll
      for (int r = 0; r < 8; ++r) {
        int m = w * 16 + ((lane >> 4) << 3) + r;
        int n = j * 16 + (lane & 15);
        sS[m * 64 + n] = s[r];
      }
    }
  }
  __syncthreads();

  // Row softmax + relative position bias (rows/cols 0..48 valid).
  if (tid < 49) {
    const int row = tid;
    float* srow = sS + row * 64;
    const int* ridx = rel_index + row * 49;
    float mx = -1e30f;
    for (int j = 0; j < 49; ++j) {
      float val = srow[j] + bias_table[ridx[j] * 12 + h];
      srow[j] = val;
      mx = fmaxf(mx, val);
    }
    float sum = 0.f;
    for (int j = 0; j < 49; ++j) {
      float p = __expf(srow[j] - mx);
      srow[j] = p;
      sum += p;
    }
    float inv = 1.f / sum;
    __bf16* prow = sP + row * 64;
    for (int j = 0; j < 49; ++j) prow[j] = (__bf16)(srow[j] * inv);
    for (int j = 49; j < 64; ++j) prow[j] = (__bf16)0.f;
  } else if (tid < 64) {
    __bf16* prow = sP + tid * 64;
    for (int j = 0; j < 64; ++j) prow[j] = (__bf16)0.f;
  }
  __syncthreads();

  // O = P V : rows 16w..16w+15, 2 x 16-col tiles, K-dim 64 (2 steps).
  v8f o[2] = {};
#pragma unroll
  for (int kk = 0; kk < 2; ++kk) {
    v16bf ap = load_frag_a(sP + w * 16 * 64 + kk * 32, 64, lane);
#pragma unroll
    for (int j = 0; j < 2; ++j) {
      v16bf bv = load_frag_b_kn(sV + kk * 32 * 32 + j * 16, 32, lane);
      o[j] = wmma_bf16(ap, bv, o[j]);
    }
  }
#pragma unroll
  for (int j = 0; j < 2; ++j) {
#pragma unroll
    for (int r = 0; r < 8; ++r) {
      int tok = w * 16 + ((lane >> 4) << 3) + r;
      if (tok < 49) {
        int d = j * 16 + (lane & 15);
        aout[((size_t)b * 49 + tok) * 384 + h * 32 + d] = (__bf16)o[j][r];
      }
    }
  }
}

// ---------------------------------------------------------------------------
// Kernel 3: out = aout(bf16) @ wpb^T + proj_b (fp32).  M=100352, K=384, N=384.
// Same double-buffered async tiling as kernel 1.
// ---------------------------------------------------------------------------
__global__ void __launch_bounds__(256) proj_gemm_kernel(
    const __bf16* __restrict__ aout, const __bf16* __restrict__ wpb,
    const float* __restrict__ proj_b, float* __restrict__ out) {
  __shared__ __align__(16) __bf16 sA[2][128 * 32];
  __shared__ __align__(16) __bf16 sB[2][64 * 32];

  const int tid  = threadIdx.x;
  const int lane = tid & 31;
  const int w    = tid >> 5;
  const int wm   = w >> 1;
  const int wn   = w & 1;
  const int m0   = blockIdx.x * 128;
  const int n0   = blockIdx.y * 64;

  const int ra0 = tid >> 2,          ca0 = (tid & 3) << 3;
  const int ra1 = (tid + 256) >> 2,  ca1 = ca0;
  const int rb  = tid >> 2,          cb  = (tid & 3) << 3;

  auto issue = [&](int kt, int buf) {
    const int k0 = kt * 32;
    async_copy_b128(&sA[buf][ra0 * 32 + ca0],
                    aout + (size_t)(m0 + ra0) * 384 + k0 + ca0);
    async_copy_b128(&sA[buf][ra1 * 32 + ca1],
                    aout + (size_t)(m0 + ra1) * 384 + k0 + ca1);
    async_copy_b128(&sB[buf][rb * 32 + cb],
                    wpb + (size_t)(n0 + rb) * 384 + k0 + cb);
  };

  v8f acc[2][2] = {};

  issue(0, 0);
  for (int kt = 0; kt < 12; ++kt) {
    const int cur = kt & 1;
    if (kt + 1 < 12) { issue(kt + 1, cur ^ 1); wait_async3(); }
    else             { wait_async0(); }
    __syncthreads();
    v16bf a0 = load_frag_a(&sA[cur][(wm * 32) * 32], 32, lane);
    v16bf a1 = load_frag_a(&sA[cur][(wm * 32 + 16) * 32], 32, lane);
    v16bf b0 = load_frag_b_nk(&sB[cur][(wn * 32) * 32], 32, lane);
    v16bf b1 = load_frag_b_nk(&sB[cur][(wn * 32 + 16) * 32], 32, lane);
    acc[0][0] = wmma_bf16(a0, b0, acc[0][0]);
    acc[0][1] = wmma_bf16(a0, b1, acc[0][1]);
    acc[1][0] = wmma_bf16(a1, b0, acc[1][0]);
    acc[1][1] = wmma_bf16(a1, b1, acc[1][1]);
    __syncthreads();
  }

#pragma unroll
  for (int j = 0; j < 2; ++j) {
    int n = n0 + wn * 32 + j * 16 + (lane & 15);
    float bias = proj_b[n];
#pragma unroll
    for (int i = 0; i < 2; ++i) {
#pragma unroll
      for (int r = 0; r < 8; ++r) {
        int m = m0 + wm * 32 + i * 16 + ((lane >> 4) << 3) + r;
        out[(size_t)m * 384 + n] = acc[i][j][r] + bias;
      }
    }
  }
}

// ---------------------------------------------------------------------------
extern "C" void kernel_launch(void* const* d_in, const int* in_sizes, int n_in,
                              void* d_out, int out_size, void* d_ws, size_t ws_size,
                              hipStream_t stream) {
  const float* x        = (const float*)d_in[0];
  const float* qkv_w    = (const float*)d_in[1];
  const float* qkv_b    = (const float*)d_in[2];
  const float* proj_w   = (const float*)d_in[3];
  const float* proj_b   = (const float*)d_in[4];
  const float* bias_tab = (const float*)d_in[5];
  const int*   rel_idx  = (const int*)d_in[6];

  const size_t NX  = (size_t)100352 * 384;          // 38,535,168 (x / aout rows)
  const size_t QKV = (size_t)2048 * 12 * 49 * 32;   // 38,535,168 per tensor
  __bf16* xb   = (__bf16*)d_ws;
  __bf16* wqb  = xb + NX;                           // 1152*384
  __bf16* wpb  = wqb + (size_t)1152 * 384;          // 384*384
  __bf16* q    = wpb + (size_t)384 * 384;
  __bf16* k    = q + QKV;
  __bf16* v    = k + QKV;
  __bf16* aout = v + QKV;

  // fp32 -> bf16 staging (x: 9,633,792 float4; weights tiny).
  cvt_bf16_kernel<<<dim3(37632), 256, 0, stream>>>(x, xb, 9633792);
  cvt_bf16_kernel<<<dim3(432),   256, 0, stream>>>(qkv_w, wqb, 110592);
  cvt_bf16_kernel<<<dim3(144),   256, 0, stream>>>(proj_w, wpb, 36864);

  // M = 784*128 exact; N_qkv = 18*64; N_proj = 6*64.
  qkv_gemm_kernel<<<dim3(784, 18), 256, 0, stream>>>(xb, wqb, qkv_b, q, k, v);
  attn_kernel<<<dim3(2048 * 12), 128, 0, stream>>>(q, k, v, bias_tab, rel_idx, aout);
  proj_gemm_kernel<<<dim3(784, 6), 256, 0, stream>>>(aout, wpb, proj_b,
                                                     (float*)d_out);
}
